// EnhancedMultiHeadAttention_14431090115033
// MI455X (gfx1250) — matrix operand order repaired and checked by reading, not compiled, not run on hardware
//
#include <hip/hip_runtime.h>
#include <hip/hip_bf16.h>
#include <math.h>

typedef __bf16 bf16_t;
typedef __attribute__((ext_vector_type(16))) __bf16 v16bf;
typedef __attribute__((ext_vector_type(8)))  float  v8f;

union FragAB { v16bf v; unsigned int u[8]; };

#define EMB   1024
#define HEADS 16
#define HDIM  64
#define SEQ   1024
#define BATCH 4
#define MROWS 4096   // BATCH*SEQ
#define RELMAX 1998  // 2*MAX_LEN-2

static __device__ __forceinline__ int iclamp(int v, int lo, int hi) {
  return v < lo ? lo : (v > hi ? hi : v);
}

static __device__ __forceinline__ float half_reduce_max(float v) {
#pragma unroll
  for (int off = 1; off < 16; off <<= 1)
    v = fmaxf(v, __shfl_xor(v, off, 32));
  return v;
}
static __device__ __forceinline__ float half_reduce_sum(float v) {
#pragma unroll
  for (int off = 1; off < 16; off <<= 1)
    v += __shfl_xor(v, off, 32);
  return v;
}

// ---------------------------------------------------------------------------
// Prep kernels: f32 -> bf16 conversion, and weight transpose+convert
// ---------------------------------------------------------------------------
__global__ void cvt_bf16_kernel(const float* __restrict__ in,
                                bf16_t* __restrict__ out, int n) {
  int i = blockIdx.x * blockDim.x + threadIdx.x;
  if (i < n) out[i] = (bf16_t)in[i];
}

// W is [K=1024][N=1024] row-major f32; write Wt[n][k] bf16.
__global__ void transpose_cvt_kernel(const float* __restrict__ W,
                                     bf16_t* __restrict__ Wt) {
  int idx = blockIdx.x * blockDim.x + threadIdx.x;   // 1M threads
  int k = idx >> 10;
  int n = idx & (EMB - 1);
  Wt[(size_t)n * EMB + k] = (bf16_t)W[(size_t)k * EMB + n];
}

// ---------------------------------------------------------------------------
// Generic bf16 WMMA GEMM: Y[m][n] = sum_k A[m][k] * Bt[n][k] + bias[n]
// A: [MROWS x 1024] bf16 row-major.  Bt: [1024 x 1024] bf16 (pre-transposed).
// One wave computes a 32(M) x 64(N) tile: 2 A-fragments reused across 4
// B-fragments -> 8 independent WMMA chains per k-step (hides the bf16
// WMMA->WMMA RAW hazard), 6 loaded dwords per WMMA.
// MODE 0: write bf16 to [B,H,N,D]   (Q, K)
// MODE 1: write bf16 to [B,H,D,N]   (V, transposed for att@V B-fragments)
// MODE 2: write f32  to [M, 1024]   (final projection -> d_out)
// ---------------------------------------------------------------------------
template <int MODE>
__global__ __launch_bounds__(256) void wmma_gemm_kernel(
    const bf16_t* __restrict__ A, const bf16_t* __restrict__ Bt,
    const float* __restrict__ bias, float* __restrict__ outF,
    bf16_t* __restrict__ outB) {
  const int K = EMB;
  int wave  = (blockIdx.x * blockDim.x + threadIdx.x) >> 5;
  int lane  = threadIdx.x & 31;
  int laneL = lane & 15;
  int khalf = lane & 16;     // B-fragment K base (0 or 16)
  int abase = khalf;         // A-fragment K base (0 or 16)
  int tn = wave & 15;        // 16 column tiles of 64
  int tm = wave >> 4;        // 128 row tiles of 32
  int m0 = tm << 5;
  int n0 = tn << 6;

  v8f acc[2][4] = { { {}, {}, {}, {} }, { {}, {}, {}, {} } };
  const unsigned int* ap0 =
      (const unsigned int*)(A + (size_t)(m0 + laneL) * K + abase);
  const unsigned int* ap1 =
      (const unsigned int*)(A + (size_t)(m0 + 16 + laneL) * K + abase);

  for (int kk = 0; kk < K; kk += 32) {
    FragAB a0, a1;
    {
      const unsigned int* p0 = ap0 + (kk >> 1);
      const unsigned int* p1 = ap1 + (kk >> 1);
#pragma unroll
      for (int i = 0; i < 8; ++i) {
        int di = i;
        a0.u[i] = p0[di];
        a1.u[i] = p1[di];
      }
    }
#pragma unroll
    for (int t = 0; t < 4; ++t) {
      FragAB b;
      const unsigned int* bp =
          (const unsigned int*)(Bt + (size_t)(n0 + t * 16 + laneL) * K + kk + khalf);
#pragma unroll
      for (int i = 0; i < 8; ++i) b.u[i] = bp[i];
      acc[0][t] = __builtin_amdgcn_wmma_f32_16x16x32_bf16(
          false, a0.v, false, b.v, (short)0, acc[0][t], false, false);
      acc[1][t] = __builtin_amdgcn_wmma_f32_16x16x32_bf16(
          false, a1.v, false, b.v, (short)0, acc[1][t], false, false);
    }
  }

#pragma unroll
  for (int g = 0; g < 2; ++g) {
    int mbase = m0 + g * 16 + (khalf >> 1);  // C/D layout: lanes>=16 -> rows +8
#pragma unroll
    for (int t = 0; t < 4; ++t) {
      int col = n0 + t * 16 + laneL;
      float bv = bias[col];
#pragma unroll
      for (int r = 0; r < 8; ++r) {
        float val = acc[g][t][r] + bv;
        int row = mbase + r;
        if (MODE == 2) {
          outF[(size_t)row * EMB + col] = val;
        } else {
          int bb  = row >> 10;
          int tok = row & (SEQ - 1);
          int h   = col >> 6;
          int d   = col & (HDIM - 1);
          size_t idx;
          if (MODE == 0)
            idx = (((size_t)(bb * HEADS + h)) * SEQ + tok) * HDIM + d;
          else
            idx = (((size_t)(bb * HEADS + h)) * HDIM + d) * SEQ + tok;
          outB[idx] = (bf16_t)val;
        }
      }
    }
  }
}

// ---------------------------------------------------------------------------
// Flash attention: one wave per (b,h, 16-query block).
// Q,K: [B,H,N,D] bf16. V: [B,H,D,N] bf16. ctx out: [B,N,C] bf16.
// ---------------------------------------------------------------------------
__global__ __launch_bounds__(32) void flash_attn_kernel(
    const bf16_t* __restrict__ Qm, const bf16_t* __restrict__ Km,
    const bf16_t* __restrict__ Vm, const float* __restrict__ bias_table,
    const float* __restrict__ temperature, bf16_t* __restrict__ ctx) {
  __shared__ __align__(16) bf16_t plds[16 * 32];

  int wg = blockIdx.x;
  int qb = wg & 63;              // SEQ/16 = 64 query blocks
  int bh = wg >> 6;              // b*HEADS + h
  int h  = bh & (HEADS - 1);
  int b  = bh >> 4;
  int lane  = threadIdx.x & 31;
  int laneL = lane & 15;
  int khalf = lane & 16;
  int abase = khalf;
  int mloc  = khalf >> 1;        // C-layout row offset of this lane half
  int q0 = qb << 4;

  float tv = temperature[0];
  tv = fminf(fmaxf(tv, 0.1f), 10.0f);
  float invScale = 1.0f / (32.0f * tv);   // 1/(sqrt(1024)*temp)

  const bf16_t* qBase = Qm + (size_t)bh * SEQ * HDIM;
  const bf16_t* kBase = Km + (size_t)bh * SEQ * HDIM;
  const bf16_t* vBase = Vm + (size_t)bh * HDIM * SEQ;

  // Q fragments for d-windows [0,32) and [32,64), resident in registers.
  FragAB aq[2];
  {
    const unsigned int* qp =
        (const unsigned int*)(qBase + (size_t)(q0 + laneL) * HDIM + abase);
#pragma unroll
    for (int dw = 0; dw < 2; ++dw)
#pragma unroll
      for (int i = 0; i < 8; ++i)
        aq[dw].u[i] = qp[dw * 16 + i];
  }

  float mrun[8], lrun[8];
  v8f o[4] = { {}, {}, {}, {} };
#pragma unroll
  for (int r = 0; r < 8; ++r) { mrun[r] = -3.0e38f; lrun[r] = 0.0f; }

  for (int jj = 0; jj < SEQ; jj += 32) {
    // ---- S = Q K^T for 32 keys (two 16x16 tiles) ----
    v8f s0 = {}, s1 = {};
#pragma unroll
    for (int dw = 0; dw < 2; ++dw) {
      FragAB kf;
      const unsigned int* kp = (const unsigned int*)(
          kBase + (size_t)(jj + laneL) * HDIM + dw * 32 + khalf);
#pragma unroll
      for (int i = 0; i < 8; ++i) kf.u[i] = kp[i];
      s0 = __builtin_amdgcn_wmma_f32_16x16x32_bf16(
          false, aq[dw].v, false, kf.v, (short)0, s0, false, false);
    }
#pragma unroll
    for (int dw = 0; dw < 2; ++dw) {
      FragAB kf;
      const unsigned int* kp = (const unsigned int*)(
          kBase + (size_t)(jj + 16 + laneL) * HDIM + dw * 32 + khalf);
#pragma unroll
      for (int i = 0; i < 8; ++i) kf.u[i] = kp[i];
      s1 = __builtin_amdgcn_wmma_f32_16x16x32_bf16(
          false, aq[dw].v, false, kf.v, (short)0, s1, false, false);
    }

    // ---- bias + scale + online softmax over the 32-key block ----
    int key0 = jj + laneL;
    float p0[8], p1[8], alpha[8];
#pragma unroll
    for (int r = 0; r < 8; ++r) {
      int qIdx = q0 + mloc + r;
      int rel0 = iclamp(qIdx - key0 + (SEQ - 1), 0, RELMAX);
      int rel1 = iclamp(qIdx - key0 - 16 + (SEQ - 1), 0, RELMAX);
      float sc0 = (s0[r] + bias_table[rel0 * HEADS + h]) * invScale;
      float sc1 = (s1[r] + bias_table[rel1 * HEADS + h]) * invScale;
      float vmax = half_reduce_max(fmaxf(sc0, sc1));
      float mnew = fmaxf(mrun[r], vmax);
      float e0 = __expf(sc0 - mnew);
      float e1 = __expf(sc1 - mnew);
      float rs = half_reduce_sum(e0 + e1);
      float al = __expf(mrun[r] - mnew);
      lrun[r] = lrun[r] * al + rs;
      mrun[r] = mnew;
      alpha[r] = al;
      p0[r] = e0;
      p1[r] = e1;
    }

    // rescale running O accumulators
#pragma unroll
    for (int t = 0; t < 4; ++t)
#pragma unroll
      for (int r = 0; r < 8; ++r) o[t][r] *= alpha[r];

    // ---- re-layout P (C-layout) -> A-fragment through LDS ----
    __syncthreads();
#pragma unroll
    for (int r = 0; r < 8; ++r) {
      plds[(mloc + r) * 32 + laneL]      = (bf16_t)p0[r];
      plds[(mloc + r) * 32 + laneL + 16] = (bf16_t)p1[r];
    }
    __syncthreads();

    FragAB pf;
    {
      const unsigned int* pp =
          (const unsigned int*)(plds + laneL * 32 + abase);
#pragma unroll
      for (int i = 0; i < 8; ++i) pf.u[i] = pp[i];
    }

    // ---- O += P @ V_block (4 d-column tiles of 16) ----
#pragma unroll
    for (int t = 0; t < 4; ++t) {
      FragAB vf;
      const unsigned int* vp = (const unsigned int*)(
          vBase + (size_t)(t * 16 + laneL) * SEQ + jj + khalf);
#pragma unroll
      for (int i = 0; i < 8; ++i) vf.u[i] = vp[i];
      o[t] = __builtin_amdgcn_wmma_f32_16x16x32_bf16(
          false, pf.v, false, vf.v, (short)0, o[t], false, false);
    }
  }

  // ---- normalize and write context [B,N,C] bf16 ----
  float inv[8];
#pragma unroll
  for (int r = 0; r < 8; ++r) inv[r] = 1.0f / lrun[r];
  size_t rowBase = ((size_t)b * SEQ + q0 + mloc) * EMB;
#pragma unroll
  for (int t = 0; t < 4; ++t) {
    int col = h * HDIM + t * 16 + laneL;
#pragma unroll
    for (int r = 0; r < 8; ++r)
      ctx[rowBase + (size_t)r * EMB + col] = (bf16_t)(o[t][r] * inv[r]);
  }
}

// ---------------------------------------------------------------------------
// Host launch
// ---------------------------------------------------------------------------
extern "C" void kernel_launch(void* const* d_in, const int* in_sizes, int n_in,
                              void* d_out, int out_size, void* d_ws, size_t ws_size,
                              hipStream_t stream) {
  (void)in_sizes; (void)n_in; (void)out_size; (void)ws_size;
  const float* x           = (const float*)d_in[0];
  const float* Wq          = (const float*)d_in[1];
  const float* bq          = (const float*)d_in[2];
  const float* Wk          = (const float*)d_in[3];
  const float* bk          = (const float*)d_in[4];
  const float* Wv          = (const float*)d_in[5];
  const float* bv          = (const float*)d_in[6];
  const float* Wp          = (const float*)d_in[7];
  const float* bp          = (const float*)d_in[8];
  const float* bias_table  = (const float*)d_in[9];
  const float* temperature = (const float*)d_in[10];
  float* out = (float*)d_out;

  char* w = (char*)d_ws;
  bf16_t* xbf = (bf16_t*)w; w += (size_t)MROWS * EMB * 2;
  bf16_t* WqT = (bf16_t*)w; w += (size_t)EMB * EMB * 2;
  bf16_t* WkT = (bf16_t*)w; w += (size_t)EMB * EMB * 2;
  bf16_t* WvT = (bf16_t*)w; w += (size_t)EMB * EMB * 2;
  bf16_t* WpT = (bf16_t*)w; w += (size_t)EMB * EMB * 2;
  bf16_t* Qb  = (bf16_t*)w; w += (size_t)MROWS * EMB * 2;
  bf16_t* Kb  = (bf16_t*)w; w += (size_t)MROWS * EMB * 2;
  bf16_t* Vb  = (bf16_t*)w; w += (size_t)MROWS * EMB * 2;
  bf16_t* ctx = (bf16_t*)w; w += (size_t)MROWS * EMB * 2;

  // 1) stage inputs as bf16 (weights transposed so B-fragments are contiguous)
  cvt_bf16_kernel<<<(MROWS * EMB) / 256, 256, 0, stream>>>(x, xbf, MROWS * EMB);
  transpose_cvt_kernel<<<(EMB * EMB) / 256, 256, 0, stream>>>(Wq, WqT);
  transpose_cvt_kernel<<<(EMB * EMB) / 256, 256, 0, stream>>>(Wk, WkT);
  transpose_cvt_kernel<<<(EMB * EMB) / 256, 256, 0, stream>>>(Wv, WvT);
  transpose_cvt_kernel<<<(EMB * EMB) / 256, 256, 0, stream>>>(Wp, WpT);

  // 2) QKV projections (WMMA GEMMs), outputs in attention-friendly layouts
  //    32x64 tile per wave -> 2048 waves -> 256 blocks of 8 waves
  wmma_gemm_kernel<0><<<256, 256, 0, stream>>>(xbf, WqT, bq, nullptr, Qb);
  wmma_gemm_kernel<0><<<256, 256, 0, stream>>>(xbf, WkT, bk, nullptr, Kb);
  wmma_gemm_kernel<1><<<256, 256, 0, stream>>>(xbf, WvT, bv, nullptr, Vb);

  // 3) flash attention with relative-position bias + temperature softmax
  flash_attn_kernel<<<BATCH * HEADS * (SEQ / 16), 32, 0, stream>>>(
      Qb, Kb, Vb, bias_table, temperature, ctx);

  // 4) output projection -> f32 d_out
  wmma_gemm_kernel<2><<<256, 256, 0, stream>>>(ctx, WpT, bp, out, nullptr);
}
